// Attention_58540404244805
// MI455X (gfx1250) — matrix operand order repaired and checked by reading, not compiled, not run on hardware
//
#include <hip/hip_runtime.h>

// Problem constants (from reference)
#define BATCH 16
#define CCH   512
#define TT    1024      // H*W = 32*32
#define NH    8
#define CH    64        // CCH / NH
#define GROUPS 32
#define CPG   16        // CCH / GROUPS
#define EPSV  1e-5f

typedef __attribute__((ext_vector_type(16))) __bf16       v16bf;
typedef __attribute__((ext_vector_type(8)))  float        v8f;
typedef __attribute__((ext_vector_type(8)))  unsigned int v8u;

static __device__ inline v16bf as_v16bf(v8u x) {
  return __builtin_bit_cast(v16bf, x);
}

static __device__ inline v8f wmma_bf16(v8u a, v8u b, v8f c) {
  return __builtin_amdgcn_wmma_f32_16x16x32_bf16(
      /*neg_a=*/false, as_v16bf(a), /*neg_b=*/false, as_v16bf(b),
      /*c_mod=*/(short)0, c, /*reuse_a=*/false, /*reuse_b=*/false);
}

// fp32 -> bf16 round-to-nearest-even
static __device__ inline unsigned short f2bf(float f) {
  unsigned int u = __builtin_bit_cast(unsigned int, f);
  unsigned int r = u + 0x7FFFu + ((u >> 16) & 1u);
  return (unsigned short)(r >> 16);
}

// A-fragment k order chosen here (not AMD's table order): per lane (m = L&15, half = L>>4),
// VGPR v holds K pair at half*16 + 2v (contiguous, matching B fills).
static __device__ inline int a_koff(int v, int half) {
  return half * 16 + v * 2;
}

// ---------------------------------------------------------------------------
// Kernel 0: convert fp32 weights to bf16 workspace copies
// ---------------------------------------------------------------------------
__global__ void cvt_weights_kernel(const float* __restrict__ qkv_w,
                                   const float* __restrict__ proj_w,
                                   unsigned short* __restrict__ qkv_w_bf,
                                   unsigned short* __restrict__ proj_w_bf) {
  int i = blockIdx.x * blockDim.x + threadIdx.x;
  if (i < 3 * CCH * CCH) qkv_w_bf[i] = f2bf(qkv_w[i]);
  if (i < CCH * CCH)     proj_w_bf[i] = f2bf(proj_w[i]);
}

// ---------------------------------------------------------------------------
// Kernel 1: GroupNorm, output t-major bf16: hT[b][t][c]
// ---------------------------------------------------------------------------
__global__ void __launch_bounds__(256)
groupnorm_kernel(const float* __restrict__ x, const float* __restrict__ w,
                 const float* __restrict__ bias, unsigned short* __restrict__ hT) {
  __shared__ float sbuf[256], sbuf2[256];
  int bg = blockIdx.x;
  int b = bg >> 5, g = bg & 31;
  const float* xg = x + ((size_t)b * CCH + (size_t)g * CPG) * TT;
  const int N = CPG * TT;

  float s = 0.f, s2 = 0.f;
  for (int i = threadIdx.x; i < N; i += blockDim.x) {
    float v = xg[i];
    s += v; s2 += v * v;
  }
  sbuf[threadIdx.x] = s; sbuf2[threadIdx.x] = s2;
  __syncthreads();
  for (int stride = 128; stride > 0; stride >>= 1) {
    if ((int)threadIdx.x < stride) {
      sbuf[threadIdx.x]  += sbuf[threadIdx.x + stride];
      sbuf2[threadIdx.x] += sbuf2[threadIdx.x + stride];
    }
    __syncthreads();
  }
  float mean = sbuf[0] / (float)N;
  float var  = sbuf2[0] / (float)N - mean * mean;
  float rstd = rsqrtf(var + EPSV);

  for (int i = threadIdx.x; i < N; i += blockDim.x) {
    int c = g * CPG + i / TT;
    int t = i % TT;
    float v = (xg[i] - mean) * rstd * w[c] + bias[c];
    hT[((size_t)b * TT + t) * CCH + c] = f2bf(v);
  }
}

// ---------------------------------------------------------------------------
// Fragment fetch helpers for the dense GEMMs (A row-major bf16, B t-major
// bf16 with contiguous K pairs).
// ---------------------------------------------------------------------------
static __device__ inline void fetch_frags(const unsigned short* __restrict__ arow,
                                          const unsigned short* const brow[4],
                                          int k, int half,
                                          v8u& a, v8u b[4]) {
#pragma unroll
  for (int v = 0; v < 8; ++v)
    a[v] = *(const unsigned int*)(arow + k + a_koff(v, half));
#pragma unroll
  for (int nt = 0; nt < 4; ++nt)
#pragma unroll
    for (int v = 0; v < 8; ++v)
      b[nt][v] = *(const unsigned int*)(brow[nt] + k + half * 16 + v * 2);
}

// ---------------------------------------------------------------------------
// Kernel 2: QKV GEMM  (o,t) = qkv_w(o,c) @ h(c,t) + bias, split/scatter q,k,v
//   Ping-pong double-buffered K loop (unrolled x2, no fragment copies).
//   qT,kT : [bh][t][64] bf16 (pre-scaled by 1/ch^0.25)
//   vv    : [bh][64][T] bf16
// ---------------------------------------------------------------------------
__global__ void __launch_bounds__(128)
qkv_gemm_kernel(const unsigned short* __restrict__ Wq,
                const float* __restrict__ qkv_b,
                const unsigned short* __restrict__ hT,
                unsigned short* __restrict__ qT,
                unsigned short* __restrict__ kT,
                unsigned short* __restrict__ vv) {
  int nt0 = blockIdx.x * 64;          // t tile
  int mt0 = blockIdx.y * 64;          // o tile
  int b   = blockIdx.z;
  int wave = threadIdx.x >> 5, lane = threadIdx.x & 31;
  int half = lane >> 4, l15 = lane & 15;
  int m0 = mt0 + wave * 16;

  const unsigned short* arow = Wq + (size_t)(m0 + l15) * CCH;
  const unsigned short* brow[4];
#pragma unroll
  for (int nt = 0; nt < 4; ++nt)
    brow[nt] = hT + ((size_t)b * TT + (nt0 + nt * 16 + l15)) * CCH;

  v8f acc[4] = {v8f{}, v8f{}, v8f{}, v8f{}};
  v8u a0, b0[4], a1, b1[4];

  fetch_frags(arow, brow, 0, half, a0, b0);
  for (int k0 = 0; k0 < CCH - 64; k0 += 64) {
    fetch_frags(arow, brow, k0 + 32, half, a1, b1);
#pragma unroll
    for (int nt = 0; nt < 4; ++nt) acc[nt] = wmma_bf16(a0, b0[nt], acc[nt]);
    fetch_frags(arow, brow, k0 + 64, half, a0, b0);
#pragma unroll
    for (int nt = 0; nt < 4; ++nt) acc[nt] = wmma_bf16(a1, b1[nt], acc[nt]);
  }
  fetch_frags(arow, brow, CCH - 32, half, a1, b1);
#pragma unroll
  for (int nt = 0; nt < 4; ++nt) acc[nt] = wmma_bf16(a0, b0[nt], acc[nt]);
#pragma unroll
  for (int nt = 0; nt < 4; ++nt) acc[nt] = wmma_bf16(a1, b1[nt], acc[nt]);

  const float qk_scale = 0.35355339059327373f;  // 1/sqrt(sqrt(64))
#pragma unroll
  for (int nt = 0; nt < 4; ++nt) {
    int t = nt0 + nt * 16 + l15;
#pragma unroll
    for (int v = 0; v < 8; ++v) {
      int o = m0 + v + 8 * half;
      float val = acc[nt][v] + qkv_b[o];
      int hh = o / (3 * CH);            // head
      int j  = o % (3 * CH);            // 0..191 within head: q|k|v
      int bh = b * NH + hh;
      if (j < CH)
        qT[((size_t)bh * TT + t) * CH + j] = f2bf(val * qk_scale);
      else if (j < 2 * CH)
        kT[((size_t)bh * TT + t) * CH + (j - CH)] = f2bf(val * qk_scale);
      else
        vv[((size_t)bh * CH + (j - 2 * CH)) * TT + t] = f2bf(val);
    }
  }
}

// ---------------------------------------------------------------------------
// Kernel 3: flash attention, per (bh, 64-row t tile); 4 waves x 16 rows.
//   S = qT kT^T via WMMA; online softmax; P staged via LDS; O += P V.
//   V-tile fragment loads are issued before the softmax VALU work so they
//   overlap with exp/shuffle math and the LDS relayout.
//   Writes attnT[b][t][c] bf16 (c = hh*64+cc) for the proj GEMM.
// ---------------------------------------------------------------------------
__global__ void __launch_bounds__(128)
attn_kernel(const unsigned short* __restrict__ qT,
            const unsigned short* __restrict__ kT,
            const unsigned short* __restrict__ vv,
            unsigned short* __restrict__ attnT) {
  __shared__ unsigned short pbuf[4 * 16 * 64];   // per-wave 16x64 P tile, bf16
  int ttile = blockIdx.x;
  int bh    = blockIdx.y;
  int wave = threadIdx.x >> 5, lane = threadIdx.x & 31;
  int half = lane >> 4, l15 = lane & 15;
  int m0 = ttile * 64 + wave * 16;
  unsigned short* pw = pbuf + wave * 16 * 64;

  // Q A-fragments (16 rows x K=64 -> two 16x32 fragments)
  v8u qa[2];
  const unsigned short* qrow = qT + ((size_t)bh * TT + m0 + l15) * CH;
#pragma unroll
  for (int kb = 0; kb < 2; ++kb)
#pragma unroll
    for (int v = 0; v < 8; ++v)
      qa[kb][v] = *(const unsigned int*)(qrow + kb * 32 + a_koff(v, half));

  float mrow[8], lrow[8];
#pragma unroll
  for (int v = 0; v < 8; ++v) { mrow[v] = -1e30f; lrow[v] = 0.f; }
  v8f O[4] = {v8f{}, v8f{}, v8f{}, v8f{}};

  for (int s0 = 0; s0 < TT; s0 += 64) {
    // ---- S = Q K^T (16 x 64 tile) ----
    v8f S[4] = {v8f{}, v8f{}, v8f{}, v8f{}};
#pragma unroll
    for (int nt = 0; nt < 4; ++nt) {
      int s = s0 + nt * 16 + l15;
      const unsigned short* krow = kT + ((size_t)bh * TT + s) * CH;
      v8u b0, b1;
#pragma unroll
      for (int v = 0; v < 8; ++v) {
        int kk = half * 16 + v * 2;
        b0[v] = *(const unsigned int*)(krow + kk);
        b1[v] = *(const unsigned int*)(krow + 32 + kk);
      }
      S[nt] = wmma_bf16(qa[0], b0, S[nt]);
      S[nt] = wmma_bf16(qa[1], b1, S[nt]);
    }

    // ---- prefetch V B-fragments for this s-block (independent of softmax) ----
    v8u vb0[4], vb1[4];
#pragma unroll
    for (int nt = 0; nt < 4; ++nt) {
      int cc = nt * 16 + l15;
      const unsigned short* vrow = vv + ((size_t)bh * CH + cc) * TT + s0;
#pragma unroll
      for (int v = 0; v < 8; ++v) {
        int kk = half * 16 + v * 2;
        vb0[nt][v] = *(const unsigned int*)(vrow + kk);
        vb1[nt][v] = *(const unsigned int*)(vrow + 32 + kk);
      }
    }

    // ---- online softmax (row m = v + 8*half, n spread over 16-lane group) ----
#pragma unroll
    for (int v = 0; v < 8; ++v) {
      float rmax = fmaxf(fmaxf(S[0][v], S[1][v]), fmaxf(S[2][v], S[3][v]));
#pragma unroll
      for (int m = 1; m < 16; m <<= 1) rmax = fmaxf(rmax, __shfl_xor(rmax, m, 32));
      float mnew = fmaxf(mrow[v], rmax);
      float corr = __expf(mrow[v] - mnew);
      mrow[v] = mnew;
      float rsum = 0.f;
#pragma unroll
      for (int nt = 0; nt < 4; ++nt) {
        float p = __expf(S[nt][v] - mnew);
        S[nt][v] = p;
        rsum += p;
      }
#pragma unroll
      for (int m = 1; m < 16; m <<= 1) rsum += __shfl_xor(rsum, m, 32);
      lrow[v] = lrow[v] * corr + rsum;
#pragma unroll
      for (int nt = 0; nt < 4; ++nt) O[nt][v] *= corr;
    }

    // ---- stage P (C-layout) to LDS, reload in A-layout ----
#pragma unroll
    for (int nt = 0; nt < 4; ++nt)
#pragma unroll
      for (int v = 0; v < 8; ++v)
        pw[(v + 8 * half) * 64 + nt * 16 + l15] = f2bf(S[nt][v]);
    __syncthreads();

    v8u pa[2];
#pragma unroll
    for (int kb = 0; kb < 2; ++kb)
#pragma unroll
      for (int v = 0; v < 8; ++v)
        pa[kb][v] = *(const unsigned int*)(pw + l15 * 64 + kb * 32 + a_koff(v, half));

    // ---- O += P @ V ----
#pragma unroll
    for (int nt = 0; nt < 4; ++nt) {
      O[nt] = wmma_bf16(pa[0], vb0[nt], O[nt]);
      O[nt] = wmma_bf16(pa[1], vb1[nt], O[nt]);
    }
    __syncthreads();
  }

  // ---- finalize and write attnT[b][t][hh*64+cc] ----
  int b = bh >> 3, hh = bh & 7;
#pragma unroll
  for (int nt = 0; nt < 4; ++nt) {
    int cc = nt * 16 + l15;
#pragma unroll
    for (int v = 0; v < 8; ++v) {
      int t = m0 + v + 8 * half;
      float val = O[nt][v] / lrow[v];
      attnT[((size_t)b * TT + t) * CCH + hh * CH + cc] = f2bf(val);
    }
  }
}

// ---------------------------------------------------------------------------
// Kernel 4: proj GEMM + bias + residual -> fp32 out[b][o][t]
//   Same ping-pong K loop as the QKV GEMM.
// ---------------------------------------------------------------------------
__global__ void __launch_bounds__(128)
proj_gemm_kernel(const unsigned short* __restrict__ Wp,
                 const float* __restrict__ proj_b,
                 const unsigned short* __restrict__ attnT,
                 const float* __restrict__ x,
                 float* __restrict__ out) {
  int nt0 = blockIdx.x * 64;   // t tile
  int mt0 = blockIdx.y * 64;   // o tile
  int b   = blockIdx.z;
  int wave = threadIdx.x >> 5, lane = threadIdx.x & 31;
  int half = lane >> 4, l15 = lane & 15;
  int m0 = mt0 + wave * 16;

  const unsigned short* arow = Wp + (size_t)(m0 + l15) * CCH;
  const unsigned short* brow[4];
#pragma unroll
  for (int nt = 0; nt < 4; ++nt)
    brow[nt] = attnT + ((size_t)b * TT + (nt0 + nt * 16 + l15)) * CCH;

  v8f acc[4] = {v8f{}, v8f{}, v8f{}, v8f{}};
  v8u a0, b0[4], a1, b1[4];

  fetch_frags(arow, brow, 0, half, a0, b0);
  for (int k0 = 0; k0 < CCH - 64; k0 += 64) {
    fetch_frags(arow, brow, k0 + 32, half, a1, b1);
#pragma unroll
    for (int nt = 0; nt < 4; ++nt) acc[nt] = wmma_bf16(a0, b0[nt], acc[nt]);
    fetch_frags(arow, brow, k0 + 64, half, a0, b0);
#pragma unroll
    for (int nt = 0; nt < 4; ++nt) acc[nt] = wmma_bf16(a1, b1[nt], acc[nt]);
  }
  fetch_frags(arow, brow, CCH - 32, half, a1, b1);
#pragma unroll
  for (int nt = 0; nt < 4; ++nt) acc[nt] = wmma_bf16(a0, b0[nt], acc[nt]);
#pragma unroll
  for (int nt = 0; nt < 4; ++nt) acc[nt] = wmma_bf16(a1, b1[nt], acc[nt]);

#pragma unroll
  for (int nt = 0; nt < 4; ++nt) {
    int t = nt0 + nt * 16 + l15;
#pragma unroll
    for (int v = 0; v < 8; ++v) {
      int o = m0 + v + 8 * half;
      size_t idx = ((size_t)b * CCH + o) * (size_t)TT + t;
      out[idx] = x[idx] + acc[nt][v] + proj_b[o];
    }
  }
}

// ---------------------------------------------------------------------------
extern "C" void kernel_launch(void* const* d_in, const int* in_sizes, int n_in,
                              void* d_out, int out_size, void* d_ws, size_t ws_size,
                              hipStream_t stream) {
  (void)in_sizes; (void)n_in; (void)out_size; (void)ws_size;
  const float* x      = (const float*)d_in[0];
  const float* norm_w = (const float*)d_in[1];
  const float* norm_b = (const float*)d_in[2];
  const float* qkv_w  = (const float*)d_in[3];
  const float* qkv_b  = (const float*)d_in[4];
  const float* proj_w = (const float*)d_in[5];
  const float* proj_b = (const float*)d_in[6];
  float* out = (float*)d_out;

  // Workspace layout (bf16 buffers), 256B aligned
  char* ws = (char*)d_ws;
  size_t off = 0;
  auto alloc = [&](size_t bytes) {
    size_t p = off;
    off += (bytes + 255) & ~(size_t)255;
    return p;
  };
  unsigned short* hT    = (unsigned short*)(ws + alloc((size_t)BATCH * TT * CCH * 2));
  unsigned short* qT    = (unsigned short*)(ws + alloc((size_t)BATCH * NH * TT * CH * 2));
  unsigned short* kT    = (unsigned short*)(ws + alloc((size_t)BATCH * NH * TT * CH * 2));
  unsigned short* vv    = (unsigned short*)(ws + alloc((size_t)BATCH * NH * CH * TT * 2));
  unsigned short* attnT = (unsigned short*)(ws + alloc((size_t)BATCH * TT * CCH * 2));
  unsigned short* qkvWb = (unsigned short*)(ws + alloc((size_t)3 * CCH * CCH * 2));
  unsigned short* projWb= (unsigned short*)(ws + alloc((size_t)CCH * CCH * 2));

  // 0) weights -> bf16
  {
    int n = 3 * CCH * CCH;
    cvt_weights_kernel<<<(n + 255) / 256, 256, 0, stream>>>(qkv_w, proj_w, qkvWb, projWb);
  }
  // 1) groupnorm -> hT
  groupnorm_kernel<<<BATCH * GROUPS, 256, 0, stream>>>(x, norm_w, norm_b, hT);
  // 2) qkv gemm -> qT, kT, vv
  {
    dim3 grid(TT / 64, (3 * CCH) / 64, BATCH);
    qkv_gemm_kernel<<<grid, 128, 0, stream>>>(qkvWb, qkv_b, hT, qT, kT, vv);
  }
  // 3) flash attention -> attnT
  {
    dim3 grid(TT / 64, BATCH * NH);
    attn_kernel<<<grid, 128, 0, stream>>>(qT, kT, vv, attnT);
  }
  // 4) proj gemm + residual -> out
  {
    dim3 grid(TT / 64, CCH / 64, BATCH);
    proj_gemm_kernel<<<grid, 128, 0, stream>>>(projWb, proj_b, attnT, x, out);
  }
}
